// HebbianMambaLayer_1709396983863
// MI455X (gfx1250) — matrix operand order repaired and checked by reading, not compiled, not run on hardware
//
#include <hip/hip_runtime.h>
#include <hip/hip_bf16.h>

// ---------------------------------------------------------------------------
// HebbianMambaLayer for MI455X (gfx1250)
// All GEMMs: v_wmma_f32_16x16x32_bf16, bf16 operands pre-converted,
// b128 fragment loads, 16x64 C strip per wave (4 WMMA per A-frag),
// fused epilogues (residual add, bf16 slice, transposed bf16 store).
// B=4, T=1024, D=512, DI=1024, DS=16, DC=4, DR=32
// ---------------------------------------------------------------------------

#define BB 4
#define TT 1024
#define DD 512
#define DI 1024
#define DS 16
#define DR 32
#define MM (BB * TT)          // 4096 flattened rows

typedef __attribute__((ext_vector_type(16))) __bf16 v16bf;
typedef __attribute__((ext_vector_type(8)))  __bf16 v8bf;
typedef __attribute__((ext_vector_type(8)))  float  v8f;

static __device__ __forceinline__ __bf16 f2bf(float f) {
    // round-to-nearest-even f32 -> bf16
    unsigned u = __builtin_bit_cast(unsigned, f);
    u += 0x7FFFu + ((u >> 16) & 1u);
    return __builtin_bit_cast(__bf16, (unsigned short)(u >> 16));
}

static __device__ __forceinline__ v8f wmma_bf(v16bf a, v16bf b, v8f c) {
    return __builtin_amdgcn_wmma_f32_16x16x32_bf16(false, a, false, b,
                                                   (short)0, c, false, false);
}

// A frag (16x32, M x K): lane l+16h holds row m, K = {k0+8h..+7, k0+16+8h..+7}
static __device__ __forceinline__ v16bf load_a_frag(const __bf16* __restrict__ row,
                                                    int k0, int h) {
    v8bf a0 = *(const v8bf*)(row + k0 + 8 * h);
    v8bf a1 = *(const v8bf*)(row + k0 + 16 + 8 * h);
    return __builtin_shufflevector(a0, a1, 0, 1, 2, 3, 4, 5, 6, 7,
                                   8, 9, 10, 11, 12, 13, 14, 15);
}

// B frag (32x16, K x N) from a row-major (N,K) matrix row: K = {k0+8h..+7, k0+16+8h..+7}
static __device__ __forceinline__ v16bf load_b_frag(const __bf16* __restrict__ row,
                                                    int k0, int h) {
    v8bf b0 = *(const v8bf*)(row + k0 + 8 * h);
    v8bf b1 = *(const v8bf*)(row + k0 + 16 + 8 * h);
    return __builtin_shufflevector(b0, b1, 0, 1, 2, 3, 4, 5, 6, 7,
                                   8, 9, 10, 11, 12, 13, 14, 15);
}

// ---------------------------------------------------------------------------
// casts / init
// ---------------------------------------------------------------------------
__global__ void cast_flat_kernel(const float* __restrict__ s, __bf16* __restrict__ d,
                                 size_t n) {
    size_t i = (size_t)blockIdx.x * blockDim.x + threadIdx.x;
    if (i < n) d[i] = f2bf(s[i]);
}

__global__ void zero_bf_kernel(__bf16* __restrict__ p, int n) {
    int i = blockIdx.x * blockDim.x + threadIdx.x;
    if (i < n) p[i] = __builtin_bit_cast(__bf16, (unsigned short)0);
}

// ---------------------------------------------------------------------------
// RMSNorm -> bf16
// ---------------------------------------------------------------------------
__global__ void rmsnorm_kernel(const float* __restrict__ x,
                               const float* __restrict__ nw,
                               __bf16* __restrict__ h, int D) {
    const int row = blockIdx.x;
    const float* xr = x + (size_t)row * D;
    __bf16* hr = h + (size_t)row * D;
    float ss = 0.f;
    for (int i = threadIdx.x; i < D; i += blockDim.x) { float v = xr[i]; ss += v * v; }
    __shared__ float red[256];
    red[threadIdx.x] = ss;
    __syncthreads();
    for (int off = 128; off > 0; off >>= 1) {
        if ((int)threadIdx.x < off) red[threadIdx.x] += red[threadIdx.x + off];
        __syncthreads();
    }
    float inv = rsqrtf(red[0] / (float)D + 1e-5f);
    for (int i = threadIdx.x; i < D; i += blockDim.x) hr[i] = f2bf(xr[i] * inv * nw[i]);
}

// ---------------------------------------------------------------------------
// WMMA GEMM, C = A @ W^T.  A (M,K) bf16, W (N,K) bf16 row-major, batched.
// 256 thr = 8 waves; wave = 16x64 strip (4 WMMA per A frag); block = 64x128.
// Outputs (all optional):
//   Cf  : f32, ldc; if e0 != null: Cf = e0 + e1 + alpha*acc (fused residual)
//   Cb  : bf16, own ldcb, only columns < Nb (fused slice-cast)
//   Cbt : bf16 transposed (BATCH, N, tb) where batch = m0/tb (packed v8bf store)
// All N used here are multiples of 64.
// ---------------------------------------------------------------------------
__global__ void gemm_rt_bf_kernel(const __bf16* __restrict__ A,
                                  const __bf16* __restrict__ W,
                                  float* __restrict__ Cf,
                                  __bf16* __restrict__ Cb,
                                  __bf16* __restrict__ Cbt,
                                  const float* __restrict__ e0,
                                  const float* __restrict__ e1, float alpha,
                                  int M, int N, int K,
                                  int lda, int ldw, int ldc, int ldcb, int Nb, int tb,
                                  size_t strideA, size_t strideW, size_t strideC) {
    const int wave = threadIdx.x >> 5;
    const int lane = threadIdx.x & 31;
    const int l = lane & 15, h = lane >> 4;
    const int m0 = blockIdx.x * 64 + (wave & 3) * 16;
    const int n0 = blockIdx.y * 128 + (wave >> 2) * 64;
    if (m0 >= M || n0 >= N) return;                  // wave-uniform
    const __bf16* arow = A + (size_t)blockIdx.z * strideA + (size_t)(m0 + l) * lda;
    const __bf16* Wb = W + (size_t)blockIdx.z * strideW;
    const __bf16* w0 = Wb + (size_t)(n0 + l) * ldw;
    const __bf16* w1 = w0 + (size_t)16 * ldw;
    const __bf16* w2 = w0 + (size_t)32 * ldw;
    const __bf16* w3 = w0 + (size_t)48 * ldw;

    v8f acc[4] = {};
    for (int k0 = 0; k0 < K; k0 += 32) {
        v16bf a = load_a_frag(arow, k0, h);
        acc[0] = wmma_bf(a, load_b_frag(w0, k0, h), acc[0]);
        acc[1] = wmma_bf(a, load_b_frag(w1, k0, h), acc[1]);
        acc[2] = wmma_bf(a, load_b_frag(w2, k0, h), acc[2]);
        acc[3] = wmma_bf(a, load_b_frag(w3, k0, h), acc[3]);
    }

    const size_t co = (size_t)blockIdx.z * strideC + (size_t)(m0 + 8 * h) * ldc + n0 + l;
    if (Cf) {
        float* p = Cf + co;
        if (e0) {
            const float* q0 = e0 + co;
            const float* q1 = e1 + co;
#pragma unroll
            for (int j = 0; j < 4; ++j)
#pragma unroll
                for (int r = 0; r < 8; ++r) {
                    size_t o = (size_t)r * ldc + j * 16;
                    p[o] = q0[o] + q1[o] + alpha * acc[j][r];
                }
        } else {
#pragma unroll
            for (int j = 0; j < 4; ++j)
#pragma unroll
                for (int r = 0; r < 8; ++r)
                    p[(size_t)r * ldc + j * 16] = acc[j][r];
        }
    }
    if (Cb) {
        __bf16* p = Cb + (size_t)blockIdx.z * strideC +
                    (size_t)(m0 + 8 * h) * ldcb + n0 + l;
#pragma unroll
        for (int j = 0; j < 4; ++j) {
            if (n0 + j * 16 < Nb) {                  // wave-uniform
#pragma unroll
                for (int r = 0; r < 8; ++r)
                    p[(size_t)r * ldcb + j * 16] = f2bf(acc[j][r]);
            }
        }
    }
    if (Cbt) {
        const int bb = m0 / tb;
        const int mloc = m0 % tb;
        __bf16* base = Cbt + (size_t)bb * N * tb;
#pragma unroll
        for (int j = 0; j < 4; ++j) {
            __bf16* p = base + (size_t)(n0 + j * 16 + l) * tb + mloc + 8 * h;
            v8bf pk;
#pragma unroll
            for (int r = 0; r < 8; ++r) pk[r] = f2bf(acc[j][r]);
            *(v8bf*)p = pk;                           // 16B packed store
        }
    }
}

// ---------------------------------------------------------------------------
// Causal depthwise conv (DC=4) + bias + SiLU.  u = xz[..., :DI] (ld = 2*DI)
// ---------------------------------------------------------------------------
__global__ void conv_silu_kernel(const float* __restrict__ xz,
                                 const float* __restrict__ cw,
                                 const float* __restrict__ cb,
                                 float* __restrict__ us,
                                 __bf16* __restrict__ usb) {
    const size_t idx = (size_t)blockIdx.x * blockDim.x + threadIdx.x;   // B*T*DI
    const int d = (int)(idx % DI);
    const size_t bt = idx / DI;
    const int t = (int)(bt % TT);
    const size_t b = bt / TT;
    float acc = cb[d];
#pragma unroll
    for (int j = 0; j < 4; ++j) {
        int tt = t - 3 + j;
        if (tt >= 0)
            acc += cw[d * 4 + j] * xz[(b * TT + (size_t)tt) * (2 * DI) + d];
    }
    float r = acc / (1.f + __expf(-acc));   // silu
    us[idx] = r;
    usb[idx] = f2bf(r);
}

// ---------------------------------------------------------------------------
// Selective scan: thread per (b,d), 16-wide state in VGPRs; Bm/Cm via LDS.
// ---------------------------------------------------------------------------
__global__ void scan_kernel(const float* __restrict__ dtraw,
                            const float* __restrict__ usilu,
                            const float* __restrict__ dbc,
                            const float* __restrict__ xz,
                            const float* __restrict__ A_log,
                            const float* __restrict__ D_skip,
                            const float* __restrict__ dt_b,
                            __bf16* __restrict__ y) {
    const int b = blockIdx.x >> 3;                    // DI/128 = 8 blocks per batch
    const int d = (blockIdx.x & 7) * 128 + threadIdx.x;
    float Arow[DS];
#pragma unroll
    for (int n = 0; n < DS; ++n) Arow[n] = -__expf(A_log[d * DS + n]);
    const float dsk = D_skip[d];
    const float dtb = dt_b[d];
    float s[DS];
#pragma unroll
    for (int n = 0; n < DS; ++n) s[n] = 0.f;

    __shared__ float bc[16][32];                      // [t-chunk][Bm(16)|Cm(16)]
    const size_t base = (size_t)b * TT;
    for (int tc = 0; tc < TT; tc += 16) {
        __syncthreads();
        for (int i = threadIdx.x; i < 512; i += 128) {
            int tl = i >> 5, col = i & 31;
            bc[tl][col] = dbc[(base + tc + tl) * 64 + 32 + col];
        }
        __syncthreads();
        for (int tl = 0; tl < 16; ++tl) {
            const size_t idx = base + tc + tl;
            float draw = dtraw[idx * DI + d] + dtb;
            float dt = (draw > 20.f) ? draw : log1pf(__expf(draw));
            float uv = usilu[idx * DI + d];
            float du = dt * uv;
            float ysum = 0.f;
#pragma unroll
            for (int n = 0; n < DS; ++n) {
                s[n] = __expf(dt * Arow[n]) * s[n] + du * bc[tl][n];
                ysum += s[n] * bc[tl][16 + n];
            }
            float rv = xz[idx * (2 * DI) + DI + d];    // res gate
            float yv = ysum + uv * dsk;
            y[idx * DI + d] = f2bf(yv * (rv / (1.f + __expf(-rv))));
        }
    }
}

// ---------------------------------------------------------------------------
// Decay-masked scores (bf16 out): 16(t) x 64(s) strip per wave, 4 WMMA per
// Q-frag, K=512.  wk[s]=out[s-1], wk[0]=0 via per-lane pointer select into a
// zeroed row.  Fully-masked strips store zeros (wave-uniform branch).
// grid: (TT/512, TT/16, BB), block 256 = 8 waves.
// ---------------------------------------------------------------------------
__global__ void scores_kernel(const __bf16* __restrict__ outb,
                              const __bf16* __restrict__ zrow,
                              const float* __restrict__ decay_p,
                              __bf16* __restrict__ scores) {
    const int b = blockIdx.z;
    const int wave = threadIdx.x >> 5;
    const int lane = threadIdx.x & 31;
    const int l = lane & 15, h = lane >> 4;
    const int t0 = blockIdx.y * 16;
    const int s0 = blockIdx.x * 512 + wave * 64;
    const __bf16* Ob = outb + (size_t)b * TT * DD;
    __bf16* srow = scores + (size_t)b * TT * TT + (size_t)(t0 + 8 * h) * TT + s0 + l;

    if (s0 > t0) {                                   // strip fully masked (uniform)
        const __bf16 z16 = __builtin_bit_cast(__bf16, (unsigned short)0);
#pragma unroll
        for (int j = 0; j < 4; ++j)
#pragma unroll
            for (int r = 0; r < 8; ++r) srow[(size_t)r * TT + j * 16] = z16;
        return;
    }
    const __bf16* qrow = Ob + (size_t)(t0 + l) * DD;
    const __bf16* krow[4];
#pragma unroll
    for (int j = 0; j < 4; ++j) {
        int sj = s0 + j * 16 + l;
        krow[j] = (sj == 0) ? zrow : Ob + (size_t)(sj - 1) * DD;
    }

    v8f acc[4] = {};
    for (int k0 = 0; k0 < DD; k0 += 32) {
        v16bf a = load_a_frag(qrow, k0, h);
        acc[0] = wmma_bf(a, load_b_frag(krow[0], k0, h), acc[0]);
        acc[1] = wmma_bf(a, load_b_frag(krow[1], k0, h), acc[1]);
        acc[2] = wmma_bf(a, load_b_frag(krow[2], k0, h), acc[2]);
        acc[3] = wmma_bf(a, load_b_frag(krow[3], k0, h), acc[3]);
    }

    const float dec = decay_p[0];
    const float lg = -__logf(1.f + __expf(-dec));    // log(sigmoid(decay))
#pragma unroll
    for (int j = 0; j < 4; ++j)
#pragma unroll
        for (int r = 0; r < 8; ++r) {
            int t = t0 + r + 8 * h;
            int sv = s0 + j * 16 + l;
            float m = (t > sv) ? __expf((float)(t - 1 - sv) * lg) : 0.f;
            srow[(size_t)r * TT + j * 16] = f2bf(acc[j][r] * m);
        }
}

// ---------------------------------------------------------------------------
extern "C" void kernel_launch(void* const* d_in, const int* in_sizes, int n_in,
                              void* d_out, int out_size, void* d_ws, size_t ws_size,
                              hipStream_t stream) {
    const float* x        = (const float*)d_in[0];
    const float* norm_w   = (const float*)d_in[1];
    const float* in_proj  = (const float*)d_in[2];   // (2048, 512)
    const float* conv_w   = (const float*)d_in[3];   // (1024, 4)
    const float* conv_b   = (const float*)d_in[4];
    const float* x_proj   = (const float*)d_in[5];   // (64, 1024)
    const float* dt_proj  = (const float*)d_in[6];   // (1024, 32)
    const float* dt_b     = (const float*)d_in[7];
    const float* A_log    = (const float*)d_in[8];   // (1024, 16)
    const float* D_skip   = (const float*)d_in[9];
    const float* out_proj = (const float*)d_in[10];  // (512, 1024)
    const float* w_write  = (const float*)d_in[11];  // (512, 512)
    const float* w_read   = (const float*)d_in[12];  // (512, 512)
    const float* decay    = (const float*)d_in[13];  // scalar

    // ---- workspace bump allocator (256B aligned), with lifetime reuse ----
    char* wsp = (char*)d_ws;
    size_t off = 0;
    auto alloc = [&](size_t bytes) -> char* {
        char* p = wsp + off;
        off += (bytes + 255) & ~(size_t)255;
        return p;
    };
    __bf16* inproj_b  = (__bf16*)alloc((size_t)2 * DI * DD * 2);
    __bf16* xproj_b   = (__bf16*)alloc((size_t)64 * DI * 2);
    __bf16* dtproj_b  = (__bf16*)alloc((size_t)DI * DR * 2);
    __bf16* outproj_b = (__bf16*)alloc((size_t)DD * DI * 2);
    __bf16* wwrite_b  = (__bf16*)alloc((size_t)DD * DD * 2);
    __bf16* wread_b   = (__bf16*)alloc((size_t)DD * DD * 2);
    __bf16* zrow      = (__bf16*)alloc((size_t)DD * 2);
    char*   hb_region = alloc((size_t)MM * DD * 2);        // h_bf -> out0_bf
    __bf16* h_bf      = (__bf16*)hb_region;
    __bf16* out0_bf   = (__bf16*)hb_region;
    char*   xz_region = alloc((size_t)MM * 2 * DI * 4);    // xz -> {scores, vT, reads}
    float*  xz        = (float*)xz_region;
    __bf16* scores_bf = (__bf16*)xz_region;                              // 8 MB
    __bf16* v_t       = (__bf16*)(xz_region + (size_t)8 * 1024 * 1024);  // 4 MB (B,D,T)
    __bf16* reads_bf  = (__bf16*)(xz_region + (size_t)12 * 1024 * 1024); // 4 MB
    float*  usilu     = (float*)alloc((size_t)MM * DI * 4);
    char*   ub_region = alloc((size_t)MM * DI * 2);        // us_bf -> y_bf
    __bf16* us_bf     = (__bf16*)ub_region;
    __bf16* y_bf      = (__bf16*)ub_region;
    float*  dbc       = (float*)alloc((size_t)MM * 64 * 4);
    __bf16* dbc32_bf  = (__bf16*)alloc((size_t)MM * DR * 2);
    char*   dt_region = alloc((size_t)MM * DI * 4);        // dtraw -> out0 (f32)
    float*  dtraw     = (float*)dt_region;
    float*  out0      = (float*)dt_region;

    // ---- weight precasts ----
    auto castN = [&](const float* s, __bf16* d, size_t n) {
        cast_flat_kernel<<<(unsigned)((n + 255) / 256), 256, 0, stream>>>(s, d, n);
    };
    castN(in_proj,  inproj_b,  (size_t)2 * DI * DD);
    castN(x_proj,   xproj_b,   (size_t)64 * DI);
    castN(dt_proj,  dtproj_b,  (size_t)DI * DR);
    castN(out_proj, outproj_b, (size_t)DD * DI);
    castN(w_write,  wwrite_b,  (size_t)DD * DD);
    castN(w_read,   wread_b,   (size_t)DD * DD);
    zero_bf_kernel<<<2, 256, 0, stream>>>(zrow, DD);

    // 1. RMSNorm -> bf16
    rmsnorm_kernel<<<MM, 256, 0, stream>>>(x, norm_w, h_bf, DD);

    // 2. xz = h @ in_proj^T   (K=512, N=2048)
    gemm_rt_bf_kernel<<<dim3(MM / 64, 2 * DI / 128, 1), 256, 0, stream>>>(
        h_bf, inproj_b, xz, nullptr, nullptr, nullptr, nullptr, 0.f,
        MM, 2 * DI, DD, DD, DD, 2 * DI, 0, 0, 0, 0, 0, 0);

    // 3. causal depthwise conv + silu on u = xz[..., :DI]
    conv_silu_kernel<<<(MM * DI) / 256, 256, 0, stream>>>(xz, conv_w, conv_b,
                                                          usilu, us_bf);

    // 4. dbc = usilu @ x_proj^T (K=1024, N=64) + fused bf16 slice of cols :32
    gemm_rt_bf_kernel<<<dim3(MM / 64, 1, 1), 256, 0, stream>>>(
        us_bf, xproj_b, dbc, dbc32_bf, nullptr, nullptr, nullptr, 0.f,
        MM, 64, DI, DI, DI, 64, DR, DR, 0, 0, 0, 0);

    // 5. dtraw = dbc[:, :32] @ dt_proj^T   (K=32, N=1024)
    gemm_rt_bf_kernel<<<dim3(MM / 64, DI / 128, 1), 256, 0, stream>>>(
        dbc32_bf, dtproj_b, dtraw, nullptr, nullptr, nullptr, nullptr, 0.f,
        MM, DI, DR, DR, DR, DI, 0, 0, 0, 0, 0, 0);

    // 6. selective scan -> y (bf16)
    scan_kernel<<<BB * 8, 128, 0, stream>>>(dtraw, usilu, dbc, xz, A_log, D_skip,
                                            dt_b, y_bf);

    // 7. out0 = y @ out_proj^T  (dual f32 + bf16 output)
    gemm_rt_bf_kernel<<<dim3(MM / 64, DD / 128, 1), 256, 0, stream>>>(
        y_bf, outproj_b, out0, out0_bf, nullptr, nullptr, nullptr, 0.f,
        MM, DD, DI, DI, DI, DD, DD, DD, 0, 0, 0, 0);

    // 8. v = out0 @ w_write^T  -> transposed bf16 v_t (B, D, T)
    gemm_rt_bf_kernel<<<dim3(MM / 64, DD / 128, 1), 256, 0, stream>>>(
        out0_bf, wwrite_b, nullptr, nullptr, v_t, nullptr, nullptr, 0.f,
        MM, DD, DD, DD, DD, DD, 0, 0, TT, 0, 0, 0);

    // 9. decay-masked scores (bf16)
    scores_kernel<<<dim3(TT / 512, TT / 16, BB), 256, 0, stream>>>(out0_bf, zrow,
                                                                   decay, scores_bf);

    // 10. reads = scores @ v = scores @ v_t^T  (batched, K=T, contiguous frags)
    gemm_rt_bf_kernel<<<dim3(TT / 64, DD / 128, BB), 256, 0, stream>>>(
        scores_bf, v_t, nullptr, reads_bf, nullptr, nullptr, nullptr, 0.f,
        TT, DD, TT, TT, TT, DD, DD, DD, 0,
        (size_t)TT * TT, (size_t)DD * TT, (size_t)TT * DD);

    // 11. d_out = x + out0 + 0.03 * (reads @ w_read^T)   (fused residual)
    gemm_rt_bf_kernel<<<dim3(MM / 64, DD / 128, 1), 256, 0, stream>>>(
        reads_bf, wread_b, (float*)d_out, nullptr, nullptr, x, out0, 0.03f,
        MM, DD, DD, DD, DD, DD, 0, 0, 0, 0, 0, 0);
}
